// RGCN_90374701843212
// MI455X (gfx1250) — matrix operand order repaired and checked by reading, not compiled, not run on hardware
//
#include <hip/hip_runtime.h>
#include <hip/hip_bf16.h>

typedef __attribute__((ext_vector_type(16))) __bf16 v16bf;
typedef __attribute__((ext_vector_type(8)))  __bf16 v8bf;
typedef __attribute__((ext_vector_type(4)))  __bf16 v4bf;
typedef __attribute__((ext_vector_type(8)))  float  v8f;

constexpr int NNODES = 10000;
constexpr int NREL   = 16;
constexpr int NBASES = 30;
constexpr int NEDGES = 160000;
constexpr int DIN    = 256;
constexpr int DHID   = 256;
constexpr int DOUT   = 128;
constexpr int KDIM   = 256;          // contraction dim is 256 for all GEMMs here
constexpr int MTILES = NNODES / 16;  // 625, exact

// ---------------------------------------------------------------------------
// A-fragment loader (bf16, wave32), row-major A [M][K].
// CDNA5 ISA 7.12.2 16-bit A 16x32:
//   lanes 0-15  row M=lane   : v0-3 = K 0..7,  v4-7 = K 16..23
//   lanes 16-31 row M=lane-16: v0-3 = K 8..15, v4-7 = K 24..31
// ---------------------------------------------------------------------------
static __device__ __forceinline__ v16bf load_a_frag(const __bf16* __restrict__ A,
                                                    int m0, int k0, int lane) {
  const int half = lane >> 4, r = lane & 15;
  const __bf16* pa = A + (size_t)(m0 + r) * KDIM + (k0 + half * 8);
  v8bf lo = *reinterpret_cast<const v8bf*>(pa);        // K: base+0..7
  v8bf hi = *reinterpret_cast<const v8bf*>(pa + 16);   // K: base+16..23
  return __builtin_shufflevector(lo, hi, 0,1,2,3,4,5,6,7,8,9,10,11,12,13,14,15);
}

// ---------------------------------------------------------------------------
// Fused GEMM kernel.  A (bf16 [NNODES][K]) x B ; Bt is B transposed, row-major
// [RCNT*NOUT][K].  Each block stages one 64-col x K strip of Bt (32 KB,
// contiguous) into LDS, shared by its 8 waves; each wave computes a 16x64
// output strip with 4 WMMA accumulators and a register-double-buffered A frag.
//   MODE 0 : Out[r] = A x W_r                       -> bf16 xw
//   MODE 1 : relu(A x root + agg + bias)            -> bf16 hidden
//   MODE 2 : A x root + agg + bias                  -> f32 final output
// B fragment (lane = column N; lanes 0-15 K 0..7,16..23; lanes 16-31 K 8..15,24..31,
// matching A) read from the transposed strip as 16B pairs.
// ---------------------------------------------------------------------------
template<int NOUT, int MODE>
__global__ __launch_bounds__(256)
void rgcn_gemm(const __bf16* __restrict__ A, const __bf16* __restrict__ Bt,
               const float* __restrict__ agg, const float* __restrict__ bias,
               __bf16* __restrict__ OutB, float* __restrict__ OutF) {
  constexpr int NCHUNK = NOUT / 64;
  constexpr int MB     = (MTILES + 7) / 8;          // 79 m-blocks of 128 rows
  __shared__ __bf16 Bs[64 * KDIM];                  // 32 KB

  const int lane = threadIdx.x & 31;
  const int wib  = threadIdx.x >> 5;
  int bid = blockIdx.x;
  const int r  = bid / (NCHUNK * MB);
  bid -= r * (NCHUNK * MB);
  const int nc = bid / MB;
  const int mb = bid - nc * MB;
  const int nb = nc * 64;
  const int m0 = (mb * 8 + wib) * 16;
  const bool active = (m0 < NNODES);

  // cooperative stage: rows [nb, nb+64) x K of this relation's Bt -> LDS
  {
    const uint4* gs = reinterpret_cast<const uint4*>(Bt + ((size_t)r * NOUT + nb) * KDIM);
    uint4* ls = reinterpret_cast<uint4*>(Bs);
#pragma unroll
    for (int i = 0; i < (64 * KDIM) / (8 * 256); ++i)
      ls[i * 256 + threadIdx.x] = gs[i * 256 + threadIdx.x];
  }
  __syncthreads();

  auto ldsB = [&](int nrel, int k0) -> v16bf {
    const int half = lane >> 4, c = lane & 15;
    const __bf16* pb = &Bs[(nrel + c) * KDIM + k0 + half * 8];
    v8bf lo = *reinterpret_cast<const v8bf*>(pb);
    v8bf hi = *reinterpret_cast<const v8bf*>(pb + 16);
    return __builtin_shufflevector(lo, hi, 0,1,2,3,4,5,6,7,8,9,10,11,12,13,14,15);
  };

  v8f c0 = {}, c1 = {}, c2 = {}, c3 = {};
  v16bf aCur = {};
  if (active) aCur = load_a_frag(A, m0, 0, lane);
#pragma unroll
  for (int k0 = 0; k0 < KDIM; k0 += 32) {
    v16bf aNext = aCur;
    if (active && (k0 + 32 < KDIM)) aNext = load_a_frag(A, m0, k0 + 32, lane);  // in flight behind WMMA
    if (active) {
      v16bf b0 = ldsB( 0, k0);
      v16bf b1 = ldsB(16, k0);
      v16bf b2 = ldsB(32, k0);
      v16bf b3 = ldsB(48, k0);
      c0 = __builtin_amdgcn_wmma_f32_16x16x32_bf16(false, aCur, false, b0, (short)0, c0, false, false);
      c1 = __builtin_amdgcn_wmma_f32_16x16x32_bf16(false, aCur, false, b1, (short)0, c1, false, false);
      c2 = __builtin_amdgcn_wmma_f32_16x16x32_bf16(false, aCur, false, b2, (short)0, c2, false, false);
      c3 = __builtin_amdgcn_wmma_f32_16x16x32_bf16(false, aCur, false, b3, (short)0, c3, false, false);
    }
    aCur = aNext;
  }
  if (!active) return;

  const int half = lane >> 4, col = lane & 15;
#pragma unroll
  for (int i = 0; i < 8; ++i) {
    const size_t base = (size_t)(m0 + half * 8 + i) * NOUT + nb + col;
    if (MODE == 0) {
      __bf16* Or = OutB + (size_t)r * NNODES * NOUT;
      Or[base +  0] = (__bf16)c0[i];
      Or[base + 16] = (__bf16)c1[i];
      Or[base + 32] = (__bf16)c2[i];
      Or[base + 48] = (__bf16)c3[i];
    } else {
      float v0 = c0[i] + agg[base +  0] + bias[nb + col +  0];
      float v1 = c1[i] + agg[base + 16] + bias[nb + col + 16];
      float v2 = c2[i] + agg[base + 32] + bias[nb + col + 32];
      float v3 = c3[i] + agg[base + 48] + bias[nb + col + 48];
      if (MODE == 1) {
        OutB[base +  0] = (__bf16)fmaxf(v0, 0.f);
        OutB[base + 16] = (__bf16)fmaxf(v1, 0.f);
        OutB[base + 32] = (__bf16)fmaxf(v2, 0.f);
        OutB[base + 48] = (__bf16)fmaxf(v3, 0.f);
      } else {
        OutF[base +  0] = v0;
        OutF[base + 16] = v1;
        OutF[base + 32] = v2;
        OutF[base + 48] = v3;
      }
    }
  }
}

// ---------------------------------------------------------------------------
// Small helper kernels
// ---------------------------------------------------------------------------
__global__ void f32_to_bf16(const float* __restrict__ in, __bf16* __restrict__ out, int n) {
  int i = blockIdx.x * 256 + threadIdx.x;
  if (i < n) out[i] = (__bf16)in[i];
}

// Wt[r][o][i] = sum_b comp[r,b] * basis[b][i][o]   (pre-transposed for WMMA B)
__global__ void build_wt(const float* __restrict__ comp, const float* __restrict__ basis,
                         __bf16* __restrict__ Wt, int Kin, int Nout) {
  int idx = blockIdx.x * 256 + threadIdx.x;
  int total = NREL * Kin * Nout;
  if (idx >= total) return;
  int i = idx % Kin;
  int o = (idx / Kin) % Nout;
  int r = idx / (Kin * Nout);
  float s = 0.f;
#pragma unroll
  for (int b = 0; b < NBASES; ++b)
    s += comp[r * NBASES + b] * basis[((size_t)b * Kin + i) * Nout + o];
  Wt[idx] = (__bf16)s;
}

// Rt[o][i] = root[i][o]
__global__ void build_roott(const float* __restrict__ root, __bf16* __restrict__ Rt,
                            int Kin, int Nout) {
  int idx = blockIdx.x * 256 + threadIdx.x;
  if (idx >= Kin * Nout) return;
  int i = idx % Kin;
  int o = idx / Kin;
  Rt[idx] = (__bf16)root[(size_t)i * Nout + o];
}

// per-(dst,rel) degree (for mean) + per-dst degree (for CSR)
__global__ void edge_count(const int* __restrict__ dst, const int* __restrict__ et,
                           float* __restrict__ cntrel, int* __restrict__ cntd) {
  int e = blockIdx.x * 256 + threadIdx.x;
  if (e >= NEDGES) return;
  int d = dst[e];
  atomicAdd(&cntrel[d * NREL + et[e]], 1.0f);
  atomicAdd(&cntd[d], 1);
}

__global__ void cnt_to_inv(float* __restrict__ cnt) {
  int i = blockIdx.x * 256 + threadIdx.x;
  if (i >= NNODES * NREL) return;
  cnt[i] = 1.0f / fmaxf(cnt[i], 1.0f);
}

// single-workgroup exclusive prefix scan over cntd -> row_ptr[N+1]
__global__ __launch_bounds__(256)
void scan_rowptr(const int* __restrict__ cntd, int* __restrict__ row_ptr) {
  __shared__ int buf[2][256];
  __shared__ int running;
  const int tid = threadIdx.x;
  if (tid == 0) running = 0;
  __syncthreads();
  for (int base = 0; base < NNODES; base += 256) {
    const int i = base + tid;
    const int v = (i < NNODES) ? cntd[i] : 0;
    int sel = 0;
    buf[0][tid] = v;
    __syncthreads();
#pragma unroll
    for (int ofs = 1; ofs < 256; ofs <<= 1) {
      int t = buf[sel][tid] + ((tid >= ofs) ? buf[sel][tid - ofs] : 0);
      buf[sel ^ 1][tid] = t;
      sel ^= 1;
      __syncthreads();
    }
    const int incl = buf[sel][tid];
    const int run  = running;
    if (i < NNODES) row_ptr[i] = run + incl - v;
    __syncthreads();
    if (tid == 0) running = run + buf[sel][255];
    __syncthreads();
  }
  if (threadIdx.x == 0) row_ptr[NNODES] = running;
}

// scatter edge ids into CSR buckets
__global__ void bucket_fill(const int* __restrict__ dst, const int* __restrict__ row_ptr,
                            int* __restrict__ cursor, int* __restrict__ eids) {
  int e = blockIdx.x * 256 + threadIdx.x;
  if (e >= NEDGES) return;
  int d = dst[e];
  int pos = atomicAdd(&cursor[d], 1);
  eids[row_ptr[d] + pos] = e;
}

// one wave per destination node: gather + weighted-accumulate its incident
// edges' rows of xw into registers, write agg once. Zero feature atomics.
template<int CH>
__global__ __launch_bounds__(256)
void csr_aggregate(const __bf16* __restrict__ xw, const int* __restrict__ src,
                   const int* __restrict__ et, const int* __restrict__ eids,
                   const int* __restrict__ row_ptr, const float* __restrict__ inv,
                   float* __restrict__ agg) {
  const int n    = blockIdx.x * 8 + (threadIdx.x >> 5);
  const int lane = threadIdx.x & 31;
  if (n >= NNODES) return;
  constexpr int CPL = CH / 32;
  float acc[CPL];
#pragma unroll
  for (int j = 0; j < CPL; ++j) acc[j] = 0.f;

  const int beg = row_ptr[n], end = row_ptr[n + 1];
  for (int idx = beg; idx < end; ++idx) {
    const int e = eids[idx];
    if (idx + 1 < end) {                       // cover random-gather latency
      const int e2 = eids[idx + 1];
      __builtin_prefetch(xw + ((size_t)et[e2] * NNODES + src[e2]) * CH, 0, 0);
    }
    const int s = src[e], t = et[e];
    const float wgt = inv[n * NREL + t];
    const __bf16* p = xw + ((size_t)t * NNODES + s) * CH + lane * CPL;
    if (CPL == 8) {
      v8bf pv = *reinterpret_cast<const v8bf*>(p);
#pragma unroll
      for (int j = 0; j < 8 && j < CPL; ++j) acc[j] += (float)pv[j] * wgt;
    } else {
      v4bf pv = *reinterpret_cast<const v4bf*>(p);
#pragma unroll
      for (int j = 0; j < 4 && j < CPL; ++j) acc[j] += (float)pv[j] * wgt;
    }
  }
  float* q = agg + (size_t)n * CH + lane * CPL;
#pragma unroll
  for (int j = 0; j < CPL; ++j) q[j] = acc[j];
}

// ---------------------------------------------------------------------------
extern "C" void kernel_launch(void* const* d_in, const int* in_sizes, int n_in,
                              void* d_out, int out_size, void* d_ws, size_t ws_size,
                              hipStream_t stream) {
  (void)in_sizes; (void)n_in; (void)out_size; (void)ws_size;
  const float* x      = (const float*)d_in[0];
  const int*   ei     = (const int*)d_in[1];
  const int*   et     = (const int*)d_in[2];
  const float* basis1 = (const float*)d_in[3];
  const float* comp1  = (const float*)d_in[4];
  const float* root1  = (const float*)d_in[5];
  const float* bias1  = (const float*)d_in[6];
  const float* basis2 = (const float*)d_in[7];
  const float* comp2  = (const float*)d_in[8];
  const float* root2  = (const float*)d_in[9];
  const float* bias2  = (const float*)d_in[10];
  const int* src = ei;
  const int* dst = ei + NEDGES;

  // workspace layout (reused across layers)
  char* ws = (char*)d_ws;
  size_t off = 0;
  auto take = [&](size_t bytes) -> char* {
    off = (off + 255) & ~(size_t)255;
    char* p = ws + off;
    off += bytes;
    return p;
  };
  __bf16* xb      = (__bf16*)take((size_t)NNODES * DIN * 2);           // x in bf16
  __bf16* hb      = (__bf16*)take((size_t)NNODES * DHID * 2);          // hidden in bf16
  __bf16* Wt      = (__bf16*)take((size_t)NREL * DIN * DHID * 2);      // relation weights (reused)
  __bf16* rT      = (__bf16*)take((size_t)DIN * DHID * 2);             // root weight (reused)
  __bf16* xw      = (__bf16*)take((size_t)NREL * NNODES * DHID * 2);   // 80 MB (reused)
  float*  inv     = (float*)take((size_t)NNODES * NREL * 4);           // 1/deg per (dst,rel)
  float*  agg     = (float*)take((size_t)NNODES * DHID * 4);           // f32 accumulator (reused)
  int*    cntd    = (int*)take((size_t)NNODES * 4);                    // per-dst degree
  int*    row_ptr = (int*)take((size_t)(NNODES + 1) * 4);              // CSR row pointers
  int*    cursor  = (int*)take((size_t)NNODES * 4);                    // bucket cursors
  int*    eids    = (int*)take((size_t)NEDGES * 4);                    // CSR edge ids

  constexpr int MB = (MTILES + 7) / 8;  // 79

  // --- graph preprocessing (identical for both layers) ---
  hipMemsetAsync(inv, 0, (size_t)NNODES * NREL * 4, stream);
  hipMemsetAsync(cntd, 0, (size_t)NNODES * 4, stream);
  hipMemsetAsync(cursor, 0, (size_t)NNODES * 4, stream);
  edge_count<<<(NEDGES + 255) / 256, 256, 0, stream>>>(dst, et, inv, cntd);
  cnt_to_inv<<<(NNODES * NREL + 255) / 256, 256, 0, stream>>>(inv);
  scan_rowptr<<<1, 256, 0, stream>>>(cntd, row_ptr);
  bucket_fill<<<(NEDGES + 255) / 256, 256, 0, stream>>>(dst, row_ptr, cursor, eids);

  // --- layer 1 ---
  f32_to_bf16<<<(NNODES * DIN + 255) / 256, 256, 0, stream>>>(x, xb, NNODES * DIN);
  build_wt<<<(NREL * DIN * DHID + 255) / 256, 256, 0, stream>>>(comp1, basis1, Wt, DIN, DHID);
  build_roott<<<(DIN * DHID + 255) / 256, 256, 0, stream>>>(root1, rT, DIN, DHID);

  rgcn_gemm<DHID, 0><<<NREL * (DHID / 64) * MB, 256, 0, stream>>>(xb, Wt, nullptr, nullptr, xw, nullptr);
  csr_aggregate<DHID><<<(NNODES + 7) / 8, 256, 0, stream>>>(xw, src, et, eids, row_ptr, inv, agg);
  rgcn_gemm<DHID, 1><<<(DHID / 64) * MB, 256, 0, stream>>>(xb, rT, agg, bias1, hb, nullptr);

  // --- layer 2 ---
  build_wt<<<(NREL * DHID * DOUT + 255) / 256, 256, 0, stream>>>(comp2, basis2, Wt, DHID, DOUT);
  build_roott<<<(DHID * DOUT + 255) / 256, 256, 0, stream>>>(root2, rT, DHID, DOUT);

  rgcn_gemm<DOUT, 0><<<NREL * (DOUT / 64) * MB, 256, 0, stream>>>(hb, Wt, nullptr, nullptr, xw, nullptr);
  csr_aggregate<DOUT><<<(NNODES + 7) / 8, 256, 0, stream>>>(xw, src, et, eids, row_ptr, inv, agg);
  rgcn_gemm<DOUT, 2><<<(DOUT / 64) * MB, 256, 0, stream>>>(hb, rT, agg, bias2, nullptr, (float*)d_out);
}
